// NeuroGNN_29729763622989
// MI455X (gfx1250) — matrix operand order repaired and checked by reading, not compiled, not run on hardware
//
#include <hip/hip_runtime.h>
#include <hip/hip_bf16.h>
#include <math.h>

typedef _Float16 half_t;
typedef __attribute__((ext_vector_type(8)))  _Float16 v8h;
typedef __attribute__((ext_vector_type(16))) _Float16 v16h;
typedef __attribute__((ext_vector_type(8)))  float    v8f;

static constexpr int N_NODES = 10000;
static constexpr int E_EDGES = 80000;
static constexpr int EA      = E_EDGES + N_NODES;   // + self loops
static constexpr int T_STEPS = 6;
static constexpr int FIN     = 8;
static constexpr int F1      = 128;                 // layers 1,2 width (4 heads x 32)
static constexpr int F3      = 32;                  // layer 3 width (1 head x 32)
static constexpr int NG      = 64;                  // graphs
static constexpr float NEG_SLOPE = 0.2f;

#define CDIV(a, b) (((a) + (b) - 1) / (b))

// ---------------------------------------------------------------- utilities
__global__ void zero_f(float* __restrict__ p, int n) {
    int i = blockIdx.x * blockDim.x + threadIdx.x;
    if (i < n) p[i] = 0.f;
}

__device__ inline void atomicMaxF(float* addr, float v) {
    // monotonic-int encoding trick; valid when slot initialized to -FLT_MAX
    if (v >= 0.f) atomicMax((int*)addr, __float_as_int(v));
    else          atomicMin((unsigned int*)addr, __float_as_uint(v));
}

// ------------------------------------------------------- f16 packing kernels
// Wt[n][k] = (n<F ? Wl[k][n] : Wr[k][n-F]), zero padded to Kp. Layout [2F x Kp].
__global__ void pack_weights(const float* __restrict__ Wl, const float* __restrict__ Wr,
                             half_t* __restrict__ Wt, int K, int F, int Kp) {
    int idx = blockIdx.x * blockDim.x + threadIdx.x;
    if (idx >= 2 * F * Kp) return;
    int n = idx / Kp, k = idx % Kp;
    float v = 0.f;
    if (k < K) v = (n < F) ? Wl[k * F + n] : Wr[k * F + (n - F)];
    Wt[idx] = (half_t)v;
}

// Xh[n][k] zero padded to Kp (row major [N x Kp])
__global__ void pack_x(const float* __restrict__ X, half_t* __restrict__ Xh,
                       int K, int Kp, int nrows) {
    int idx = blockIdx.x * blockDim.x + threadIdx.x;
    if (idx >= nrows * Kp) return;
    int n = idx / Kp, k = idx % Kp;
    Xh[idx] = (k < K) ? (half_t)X[n * K + k] : (half_t)0.f;
}

// ------------------------------------------------------------- WMMA GEMM
// Out[N x Ncols] = Xh[N x Kp] * Wt^T  (Wt stored [Ncols x Kp])
// One wave -> one 16x16 tile. block = (32,4) waves, grid = (N/16, Ncols/16/4)
__global__ __launch_bounds__(128)
void gemm_wmma(const half_t* __restrict__ Xh, const half_t* __restrict__ Wt,
               float* __restrict__ Out, int Kp, int Ncols) {
    const int lane    = threadIdx.x;            // 0..31 (wave32)
    const int rowBase = blockIdx.x * 16;
    const int colBase = (blockIdx.y * 4 + threadIdx.y) * 16;
    const int mh = lane & 15;                   // M (A) / N (B) within tile
    const int hi = lane >> 4;                   // lane-half selects K split

    v8f acc = {};
    const half_t* pA = Xh + (size_t)(rowBase + mh) * Kp + hi * 8;
    const half_t* pB = Wt + (size_t)(colBase + mh) * Kp + hi * 8;

    for (int k0 = 0; k0 < Kp; k0 += 32) {
        // A 16x32 f16 fragment: elems 0..7 -> K = k0 + hi*8 + e ; 8..15 -> +16
        v8h a0 = *reinterpret_cast<const v8h*>(pA + k0);
        v8h a1 = *reinterpret_cast<const v8h*>(pA + k0 + 16);
        v16h a = __builtin_shufflevector(a0, a1, 0,1,2,3,4,5,6,7,8,9,10,11,12,13,14,15);
        // B 32x16 f16 fragment: elems 0..7 -> K = k0 + hi*8 + e ; 8..15 -> +16
        v8h b0 = *reinterpret_cast<const v8h*>(pB + k0);
        v8h b1 = *reinterpret_cast<const v8h*>(pB + k0 + 16);
        v16h b = __builtin_shufflevector(b0, b1, 0,1,2,3,4,5,6,7,8,9,10,11,12,13,14,15);
        acc = __builtin_amdgcn_wmma_f32_16x16x32_f16(false, a, false, b,
                                                     (short)0, acc, false, false);
    }
    // C/D: VGPR r -> M = r + hi*8, N = lane&15
    float* po = Out + (size_t)(rowBase + hi * 8) * Ncols + colBase + mh;
#pragma unroll
    for (int r = 0; r < 8; ++r) po[(size_t)r * Ncols] = acc[r];
}

// ------------------------------------------------------------ edge kernels
__global__ void init_layer(float* __restrict__ emax, float* __restrict__ denom,
                           float* __restrict__ agg, int nh, int nf) {
    int i = blockIdx.x * blockDim.x + threadIdx.x;
    if (i < nh) { emax[i] = -3.402823466e38f; denom[i] = 0.f; }
    if (i < nf) agg[i] = 0.f;
}

// score per (edge, head): a . leakyrelu(xl[src] + xr[dst]); segment max via atomics
__global__ void edge_scores(const int* __restrict__ src, const int* __restrict__ dst,
                            const float* __restrict__ XLXR, const float* __restrict__ att,
                            float* __restrict__ esc, float* __restrict__ emax,
                            int nH, int C, int F) {
    int idx = blockIdx.x * blockDim.x + threadIdx.x;
    if (idx >= EA * nH) return;
    int e = idx / nH, h = idx % nH;
    int s, d;
    if (e < E_EDGES) { s = src[e]; d = dst[e]; } else { s = d = e - E_EDGES; }
    const float* xl = XLXR + (size_t)s * (2 * F) + h * C;
    const float* xr = XLXR + (size_t)d * (2 * F) + F + h * C;
    const float* a  = att + h * C;
    float sc = 0.f;
    for (int c = 0; c < C; ++c) {
        float v = xl[c] + xr[c];
        v = v > 0.f ? v : NEG_SLOPE * v;
        sc += a[c] * v;
    }
    esc[idx] = sc;
    atomicMaxF(&emax[d * nH + h], sc);
}

// p = exp(e - max); segment-sum denominator
__global__ void edge_exp(const int* __restrict__ dst, float* __restrict__ esc,
                         const float* __restrict__ emax, float* __restrict__ denom, int nH) {
    int idx = blockIdx.x * blockDim.x + threadIdx.x;
    if (idx >= EA * nH) return;
    int e = idx / nH, h = idx % nH;
    int d = (e < E_EDGES) ? dst[e] : (e - E_EDGES);
    float p = __expf(esc[idx] - emax[d * nH + h]);
    esc[idx] = p;
    atomicAdd(&denom[d * nH + h], p);
}

// agg[dst] += alpha * xl[src]; thread handles 4 consecutive features (same head)
__global__ void edge_agg(const int* __restrict__ src, const int* __restrict__ dst,
                         const float* __restrict__ esc, const float* __restrict__ denom,
                         const float* __restrict__ XLXR, float* __restrict__ agg,
                         int nH, int C, int F) {
    int nQ  = F >> 2;
    int idx = blockIdx.x * blockDim.x + threadIdx.x;
    if (idx >= EA * nQ) return;
    int e = idx / nQ, f = (idx % nQ) << 2;
    int h = f / C;
    int s, d;
    if (e < E_EDGES) { s = src[e]; d = dst[e]; } else { s = d = e - E_EDGES; }
    float alpha = esc[e * nH + h] / (denom[d * nH + h] + 1e-16f);
    const float* xl = XLXR + (size_t)s * (2 * F) + f;
    float* o = agg + (size_t)d * F + f;
#pragma unroll
    for (int i = 0; i < 4; ++i) atomicAdd(&o[i], alpha * xl[i]);
}

// h = elu(agg + bias); optionally emit f16 (next layer GEMM input, K==F) and/or f32
__global__ void finalize_elu(const float* __restrict__ agg, const float* __restrict__ bias,
                             half_t* __restrict__ xh, float* __restrict__ hf, int F) {
    int idx = blockIdx.x * blockDim.x + threadIdx.x;
    if (idx >= N_NODES * F) return;
    float v = agg[idx] + bias[idx % F];
    v = v > 0.f ? v : (__expf(v) - 1.f);
    if (xh) xh[idx] = (half_t)v;
    if (hf) hf[idx] = v;
}

// ------------------------------------------------------------ pooling + GRU
__global__ void count_graphs(const int* __restrict__ batch, float* __restrict__ cnt) {
    int i = blockIdx.x * blockDim.x + threadIdx.x;
    if (i < N_NODES) atomicAdd(&cnt[batch[i]], 1.f);
}

__global__ void pool_add(const float* __restrict__ h3, const int* __restrict__ batch,
                         float* __restrict__ pooled_t) {
    int idx = blockIdx.x * blockDim.x + threadIdx.x;
    if (idx >= N_NODES * F3) return;
    int n = idx >> 5, c = idx & 31;
    atomicAdd(&pooled_t[batch[n] * F3 + c], h3[idx]);
}

__global__ __launch_bounds__(256)
void gru_head(const float* __restrict__ pooled, const float* __restrict__ cnt,
              const float* __restrict__ Wih, const float* __restrict__ Whh,
              const float* __restrict__ bih, const float* __restrict__ bhh,
              const float* __restrict__ Wc, const float* __restrict__ bc,
              float* __restrict__ out) {
    __shared__ float sh[NG * 64];   // hidden
    __shared__ float sn[NG * 64];   // new hidden
    __shared__ float sx[NG * 32];   // input step
    int tid = threadIdx.x;
    for (int i = tid; i < NG * 64; i += 256) sh[i] = 0.f;
    __syncthreads();
    for (int t = 0; t < T_STEPS; ++t) {
        for (int i = tid; i < NG * 32; i += 256) {
            float c = cnt[i >> 5]; c = c < 1.f ? 1.f : c;
            sx[i] = pooled[t * NG * 32 + i] / c;
        }
        __syncthreads();
        for (int pg = tid; pg < NG * 64; pg += 256) {
            int b = pg >> 6, j = pg & 63;
            float gr = bih[j],        hr = bhh[j];
            float gz = bih[64 + j],   hz = bhh[64 + j];
            float gn = bih[128 + j],  hn = bhh[128 + j];
            for (int c = 0; c < 32; ++c) {
                float xv = sx[b * 32 + c];
                gr += xv * Wih[j * 32 + c];
                gz += xv * Wih[(64 + j) * 32 + c];
                gn += xv * Wih[(128 + j) * 32 + c];
            }
            for (int k = 0; k < 64; ++k) {
                float hv = sh[b * 64 + k];
                hr += hv * Whh[j * 64 + k];
                hz += hv * Whh[(64 + j) * 64 + k];
                hn += hv * Whh[(128 + j) * 64 + k];
            }
            float r = 1.f / (1.f + __expf(-(gr + hr)));
            float z = 1.f / (1.f + __expf(-(gz + hz)));
            float n = tanhf(gn + r * hn);
            sn[pg] = (1.f - z) * n + z * sh[pg];
        }
        __syncthreads();
        for (int i = tid; i < NG * 64; i += 256) sh[i] = sn[i];
        __syncthreads();
    }
    if (tid < NG) {
        float acc = bc[0];
        for (int j = 0; j < 64; ++j) acc += sh[tid * 64 + j] * Wc[j];
        out[tid] = acc;
    }
}

// ---------------------------------------------------------------- dispatch
extern "C" void kernel_launch(void* const* d_in, const int* in_sizes, int n_in,
                              void* d_out, int out_size, void* d_ws, size_t ws_size,
                              hipStream_t stream) {
    const float* x    = (const float*)d_in[0];
    const int* eidx   = (const int*)d_in[1];
    const int* batch  = (const int*)d_in[2];
    const float* Wl1  = (const float*)d_in[3];
    const float* Wr1  = (const float*)d_in[4];
    const float* att1 = (const float*)d_in[5];
    const float* b1   = (const float*)d_in[6];
    const float* Wl2  = (const float*)d_in[7];
    const float* Wr2  = (const float*)d_in[8];
    const float* att2 = (const float*)d_in[9];
    const float* b2   = (const float*)d_in[10];
    const float* Wl3  = (const float*)d_in[11];
    const float* Wr3  = (const float*)d_in[12];
    const float* att3 = (const float*)d_in[13];
    const float* b3   = (const float*)d_in[14];
    const float* Wih  = (const float*)d_in[15];
    const float* Whh  = (const float*)d_in[16];
    const float* bih  = (const float*)d_in[17];
    const float* bhh  = (const float*)d_in[18];
    const float* Wc   = (const float*)d_in[19];
    const float* bc   = (const float*)d_in[20];
    float* out = (float*)d_out;

    // workspace carve (256B aligned regions)
    char* wp = (char*)d_ws;
    auto carve = [&](size_t bytes) -> void* {
        void* p = (void*)wp;
        wp += (bytes + 255) & ~(size_t)255;
        return p;
    };
    half_t* Xh    = (half_t*)carve((size_t)N_NODES * 128 * sizeof(half_t));
    half_t* Wt    = (half_t*)carve((size_t)256 * 128 * sizeof(half_t));
    float* XLXR   = (float*)carve((size_t)N_NODES * 256 * sizeof(float));
    float* agg    = (float*)carve((size_t)N_NODES * 128 * sizeof(float));
    float* emax   = (float*)carve((size_t)N_NODES * 4 * sizeof(float));
    float* denom  = (float*)carve((size_t)N_NODES * 4 * sizeof(float));
    float* esc    = (float*)carve((size_t)EA * 4 * sizeof(float));
    float* h3     = (float*)carve((size_t)N_NODES * F3 * sizeof(float));
    float* pooled = (float*)carve((size_t)T_STEPS * NG * F3 * sizeof(float));
    float* cnt    = (float*)carve((size_t)NG * sizeof(float));

    const int BT = 256;
    zero_f<<<CDIV(T_STEPS * NG * F3, BT), BT, 0, stream>>>(pooled, T_STEPS * NG * F3);
    zero_f<<<CDIV(NG, BT), BT, 0, stream>>>(cnt, NG);
    count_graphs<<<CDIV(N_NODES, BT), BT, 0, stream>>>(batch, cnt);

    for (int t = 0; t < T_STEPS; ++t) {
        const float* xt = x + (size_t)t * N_NODES * FIN;
        const int* src  = eidx + (size_t)t * 2 * E_EDGES;
        const int* dst  = src + E_EDGES;

        // ---------------- layer 1: Fin=8 (Kp=32) -> F=128, H=4
        pack_x<<<CDIV(N_NODES * 32, BT), BT, 0, stream>>>(xt, Xh, FIN, 32, N_NODES);
        pack_weights<<<CDIV(2 * F1 * 32, BT), BT, 0, stream>>>(Wl1, Wr1, Wt, FIN, F1, 32);
        gemm_wmma<<<dim3(N_NODES / 16, (2 * F1 / 16) / 4), dim3(32, 4), 0, stream>>>(
            Xh, Wt, XLXR, 32, 2 * F1);
        init_layer<<<CDIV(N_NODES * F1, BT), BT, 0, stream>>>(emax, denom, agg,
                                                              N_NODES * 4, N_NODES * F1);
        edge_scores<<<CDIV(EA * 4, BT), BT, 0, stream>>>(src, dst, XLXR, att1, esc, emax, 4, 32, F1);
        edge_exp<<<CDIV(EA * 4, BT), BT, 0, stream>>>(dst, esc, emax, denom, 4);
        edge_agg<<<CDIV(EA * (F1 / 4), BT), BT, 0, stream>>>(src, dst, esc, denom, XLXR, agg, 4, 32, F1);
        finalize_elu<<<CDIV(N_NODES * F1, BT), BT, 0, stream>>>(agg, b1, Xh, nullptr, F1);

        // ---------------- layer 2: 128 -> 128, H=4
        pack_weights<<<CDIV(2 * F1 * 128, BT), BT, 0, stream>>>(Wl2, Wr2, Wt, F1, F1, 128);
        gemm_wmma<<<dim3(N_NODES / 16, (2 * F1 / 16) / 4), dim3(32, 4), 0, stream>>>(
            Xh, Wt, XLXR, 128, 2 * F1);
        init_layer<<<CDIV(N_NODES * F1, BT), BT, 0, stream>>>(emax, denom, agg,
                                                              N_NODES * 4, N_NODES * F1);
        edge_scores<<<CDIV(EA * 4, BT), BT, 0, stream>>>(src, dst, XLXR, att2, esc, emax, 4, 32, F1);
        edge_exp<<<CDIV(EA * 4, BT), BT, 0, stream>>>(dst, esc, emax, denom, 4);
        edge_agg<<<CDIV(EA * (F1 / 4), BT), BT, 0, stream>>>(src, dst, esc, denom, XLXR, agg, 4, 32, F1);
        finalize_elu<<<CDIV(N_NODES * F1, BT), BT, 0, stream>>>(agg, b2, Xh, nullptr, F1);

        // ---------------- layer 3: 128 -> 32, H=1
        pack_weights<<<CDIV(2 * F3 * 128, BT), BT, 0, stream>>>(Wl3, Wr3, Wt, F1, F3, 128);
        gemm_wmma<<<dim3(N_NODES / 16, (2 * F3 / 16) / 4), dim3(32, 4), 0, stream>>>(
            Xh, Wt, XLXR, 128, 2 * F3);
        init_layer<<<CDIV(N_NODES * F3, BT), BT, 0, stream>>>(emax, denom, agg,
                                                              N_NODES * 1, N_NODES * F3);
        edge_scores<<<CDIV(EA * 1, BT), BT, 0, stream>>>(src, dst, XLXR, att3, esc, emax, 1, 32, F3);
        edge_exp<<<CDIV(EA * 1, BT), BT, 0, stream>>>(dst, esc, emax, denom, 1);
        edge_agg<<<CDIV(EA * (F3 / 4), BT), BT, 0, stream>>>(src, dst, esc, denom, XLXR, agg, 1, 32, F3);
        finalize_elu<<<CDIV(N_NODES * F3, BT), BT, 0, stream>>>(agg, b3, nullptr, h3, F3);

        // ---------------- mean pool into pooled[t]
        pool_add<<<CDIV(N_NODES * F3, BT), BT, 0, stream>>>(h3, batch, pooled + t * NG * F3);
    }

    // ---------------- GRU over T steps + linear head (one workgroup)
    gru_head<<<1, 256, 0, stream>>>(pooled, cnt, Wih, Whh, bih, bhh, Wc, bc, out);
}
